// FullUDHead_9079560864423
// MI455X (gfx1250) — compile-verified, matrix operand order checked by reading, not run on hardware
//
#include <hip/hip_runtime.h>
#include <hip/hip_bf16.h>
#include <stdint.h>
#include <stddef.h>

typedef __attribute__((ext_vector_type(16))) __bf16 v16b;
typedef __attribute__((ext_vector_type(8)))  __bf16 v8b;
typedef __attribute__((ext_vector_type(8)))  float  v8f;

#define B_    32
#define T_    256
#define D_    768
#define H_    400
#define HP    416      // recurrent K padded to 13*32
#define G4    1600     // 4*H
#define DC    800      // 2*H
#define A_    500
#define AP    512      // 501 -> padded
#define R_    100
#define RP    128      // 101 -> padded
#define NRELS 37
#define TB    (T_*B_)

__device__ __forceinline__ v8f vzero8() {
  v8f z = {0.f,0.f,0.f,0.f,0.f,0.f,0.f,0.f};
  return z;
}

// Load one 16(row) x 32(K) bf16 fragment for v_wmma_f32_16x16x32_bf16 (A or B
// operand). base -> element [row0][k0]; `stride` in elements (multiple of 8 so
// the two 16B chunks are aligned). Branch-free: callers guarantee rows exist
// (N dims are zero-padded at pack time).
__device__ __forceinline__ v16b load_frag(const __bf16* base, int stride) {
  const int lane = threadIdx.x & 31;
  const int r  = lane & 15;
  const int kb = (lane & 16) ? 8 : 0;   // low lanes: K 0..7/16..23, high: 8..15/24..31
  const __bf16* p = base + (size_t)r * stride + kb;
  v8b lo = *(const v8b*)(p);
  v8b hi = *(const v8b*)(p + 16);
  v16b f;
#pragma unroll
  for (int i = 0; i < 8; ++i) { f[i] = lo[i]; f[i + 8] = hi[i]; }
  return f;
}

__device__ __forceinline__ v8f wmma_bf(v16b a, v16b b, v8f c) {
  return __builtin_amdgcn_wmma_f32_16x16x32_bf16(false, a, false, b, (short)0, c, false, false);
}

// ---------------------------------------------------------------- pack / cast
__global__ void cast_bf16_kernel(const float* __restrict__ s, __bf16* __restrict__ d, size_t n) {
  for (size_t i = (size_t)blockIdx.x * blockDim.x + threadIdx.x; i < n;
       i += (size_t)gridDim.x * blockDim.x)
    d[i] = (__bf16)s[i];
}

// src (nreal x K) f32 -> dst (npad x K) bf16, rows >= nreal zero-filled
__global__ void pack_padrows_kernel(const float* __restrict__ s, __bf16* __restrict__ d,
                                    int nreal, int npad, int K) {
  const size_t n = (size_t)npad * K;
  for (size_t i = (size_t)blockIdx.x * blockDim.x + threadIdx.x; i < n;
       i += (size_t)gridDim.x * blockDim.x) {
    int row = (int)(i / K);
    d[i] = (row < nreal) ? (__bf16)s[i] : (__bf16)0.f;  // flat index matches (same K)
  }
}

// x (B,T,D) f32 -> xs (T,B,D) bf16
__global__ void pack_x_kernel(const float* __restrict__ x, __bf16* __restrict__ xs) {
  const size_t n = (size_t)B_ * T_ * D_;
  for (size_t i = (size_t)blockIdx.x * blockDim.x + threadIdx.x; i < n;
       i += (size_t)gridDim.x * blockDim.x) {
    size_t b = i / ((size_t)T_ * D_);
    size_t r = i % ((size_t)T_ * D_);
    size_t t = r / D_, d = r % D_;
    xs[(t * B_ + b) * D_ + d] = (__bf16)x[i];
  }
}

// nmat matrices of (G4 x H_) f32 -> (G4 x HP) bf16, K zero-padded
__global__ void pack_whh_kernel(const float* __restrict__ w, __bf16* __restrict__ o, int nmat) {
  const size_t n = (size_t)nmat * G4 * HP;
  for (size_t i = (size_t)blockIdx.x * blockDim.x + threadIdx.x; i < n;
       i += (size_t)gridDim.x * blockDim.x) {
    int k   = (int)(i % HP);
    size_t rm = i / HP;                 // mat*G4 + row
    o[i] = (k < H_) ? (__bf16)w[rm * H_ + k] : (__bf16)0.f;
  }
}

// W_arc (501,500) -> transposed padded (AP x AP): out[n*AP+k] = W[k][n]
__global__ void pack_warcT_kernel(const float* __restrict__ w, __bf16* __restrict__ o) {
  const size_t n = (size_t)AP * AP;
  for (size_t i = (size_t)blockIdx.x * blockDim.x + threadIdx.x; i < n;
       i += (size_t)gridDim.x * blockDim.x) {
    int nn = (int)(i / AP), k = (int)(i % AP);
    o[i] = (nn < A_ && k < A_ + 1) ? (__bf16)w[(size_t)k * A_ + nn] : (__bf16)0.f;
  }
}

// W_rel (37,101,101) -> transposed padded (37,RP,RP): out[o][j*RP+i] = W[o][i][j]
__global__ void pack_wrelT_kernel(const float* __restrict__ w, __bf16* __restrict__ o) {
  const size_t n = (size_t)NRELS * RP * RP;
  for (size_t i = (size_t)blockIdx.x * blockDim.x + threadIdx.x; i < n;
       i += (size_t)gridDim.x * blockDim.x) {
    int oo  = (int)(i / (RP * RP));
    int rem = (int)(i % (RP * RP));
    int j = rem / RP, ii = rem % RP;
    o[i] = (ii < R_ + 1 && j < R_ + 1)
               ? (__bf16)w[((size_t)oo * (R_ + 1) + ii) * (R_ + 1) + j]
               : (__bf16)0.f;
  }
}

// fill pad buffer: col==onecol -> 1, else 0 (real cols overwritten by GEMM later)
__global__ void init_pad_kernel(__bf16* __restrict__ d, size_t rows, int ld, int onecol) {
  const size_t n = rows * (size_t)ld;
  for (size_t i = (size_t)blockIdx.x * blockDim.x + threadIdx.x; i < n;
       i += (size_t)gridDim.x * blockDim.x) {
    int c = (int)(i % ld);
    d[i] = (c == onecol) ? (__bf16)1.f : (__bf16)0.f;
  }
}

// ---------------------------------------------------------------- generic GEMM
// out[m, ocol+n] = act( sum_k A[m,k]*Bw[n,k] + bias[n] ), bf16 out, f32 accum.
// Ping-pong software pipeline (two named fragment sets, each reloaded only
// after its WMMAs consumed it): loads overlap matrix ops with no register
// rotation copies. Bw must have >= 64*gridDim.y valid (zero-padded) rows.
// grid = (M/128, Npad/64); block = 256 (8 waves); wave w owns M-tile w, 4 N-tiles.
__global__ void __launch_bounds__(256)
gemm_bf16_kernel(const __bf16* __restrict__ A, int lda,
                 const __bf16* __restrict__ Bw, int ldb,
                 int N, int K,
                 const float* __restrict__ bias,
                 __bf16* __restrict__ out, int ldo, int ocol, int act) {
  const int lane = threadIdx.x & 31;
  const int wave = threadIdx.x >> 5;
  const int mrow = blockIdx.x * 128 + wave * 16;
  const int nbase = blockIdx.y * 64;
  const __bf16* Arow = A + (size_t)mrow * lda;
  const int S = K >> 5;   // number of 32-wide K steps (>= 2 for all our GEMMs)
  v8f acc[4];
#pragma unroll
  for (int j = 0; j < 4; ++j) acc[j] = vzero8();

  v16b afA, afB, bfA[4], bfB[4];
  afA = load_frag(Arow, lda);
#pragma unroll
  for (int j = 0; j < 4; ++j)
    bfA[j] = load_frag(Bw + (size_t)(nbase + 16 * j) * ldb, ldb);

  int s = 1;
  for (; s + 1 < S; s += 2) {
    afB = load_frag(Arow + (size_t)s * 32, lda);
#pragma unroll
    for (int j = 0; j < 4; ++j)
      bfB[j] = load_frag(Bw + (size_t)(nbase + 16 * j) * ldb + (size_t)s * 32, ldb);
#pragma unroll
    for (int j = 0; j < 4; ++j) acc[j] = wmma_bf(afA, bfA[j], acc[j]);
    afA = load_frag(Arow + (size_t)(s + 1) * 32, lda);
#pragma unroll
    for (int j = 0; j < 4; ++j)
      bfA[j] = load_frag(Bw + (size_t)(nbase + 16 * j) * ldb + (size_t)(s + 1) * 32, ldb);
#pragma unroll
    for (int j = 0; j < 4; ++j) acc[j] = wmma_bf(afB, bfB[j], acc[j]);
  }
  if (s < S) {           // one trailing pair (even S): prefetch last step, drain
    afB = load_frag(Arow + (size_t)s * 32, lda);
#pragma unroll
    for (int j = 0; j < 4; ++j)
      bfB[j] = load_frag(Bw + (size_t)(nbase + 16 * j) * ldb + (size_t)s * 32, ldb);
#pragma unroll
    for (int j = 0; j < 4; ++j) acc[j] = wmma_bf(afA, bfA[j], acc[j]);
#pragma unroll
    for (int j = 0; j < 4; ++j) acc[j] = wmma_bf(afB, bfB[j], acc[j]);
  } else {
#pragma unroll
    for (int j = 0; j < 4; ++j) acc[j] = wmma_bf(afA, bfA[j], acc[j]);
  }

  const int hi8 = (lane >> 4) * 8, nl = lane & 15;
#pragma unroll
  for (int j = 0; j < 4; ++j)
#pragma unroll
    for (int r = 0; r < 8; ++r) {
      int gn = nbase + 16 * j + nl;
      if (gn < N) {
        float v = acc[j][r] + (bias ? bias[gn] : 0.f);
        if (act == 1) v = (v > 0.f) ? v : 0.1f * v;
        out[(size_t)(mrow + hi8 + r) * ldo + ocol + gn] = (__bf16)v;
      }
    }
}

// ---------------------------------------------------------------- LSTM layer
// Persistent: one block per direction (grid=2), 800 threads = 25 waves.
// Wave w owns h columns [16w,16w+16). gates = P[t] + h @ Whh^T, elementwise,
// h kept in LDS (bf16, K padded to HP), c in VGPRs. Writes y into ycat[t*B,:].
__global__ void __launch_bounds__(800)
lstm_layer_kernel(const __bf16* __restrict__ P,    // [2][TB][G4] (bias pre-added)
                  const __bf16* __restrict__ Whh,  // [2][G4][HP]
                  const int* __restrict__ lengths, // [B]
                  __bf16* __restrict__ ycat) {     // [TB][DC], this dir at col dir*H_
  __shared__ __bf16 h_lds[B_ * HP];
  const int dir = blockIdx.x;
  const __bf16* Pd = P + (size_t)dir * TB * G4;
  const __bf16* Wd = Whh + (size_t)dir * G4 * HP;
  const int tid = threadIdx.x, lane = tid & 31, wave = tid >> 5;
  const int nl = lane & 15, hi8 = (lane >> 4) * 8;
  const int ncol = wave * 16;           // < 400 always (25 waves)

  for (int i = tid; i < B_ * HP; i += blockDim.x) h_lds[i] = (__bf16)0.f;
  __syncthreads();

  int len[2][8];
  float cst[2][8];
#pragma unroll
  for (int mt = 0; mt < 2; ++mt)
#pragma unroll
    for (int r = 0; r < 8; ++r) {
      len[mt][r] = lengths[mt * 16 + hi8 + r];
      cst[mt][r] = 0.f;
    }

  for (int step = 0; step < T_; ++step) {
    const int t = dir ? (T_ - 1 - step) : step;
    v8f acc[2][4];
#pragma unroll
    for (int mt = 0; mt < 2; ++mt)
#pragma unroll
      for (int g = 0; g < 4; ++g) {
#pragma unroll
        for (int r = 0; r < 8; ++r) {
          int gm = mt * 16 + hi8 + r;
          acc[mt][g][r] = (float)Pd[((size_t)t * B_ + gm) * G4 + g * H_ + ncol + nl];
        }
      }
    for (int k0 = 0; k0 < HP; k0 += 32) {
      v16b a0 = load_frag(h_lds + k0, HP);
      v16b a1 = load_frag(h_lds + 16 * HP + k0, HP);
#pragma unroll
      for (int g = 0; g < 4; ++g) {
        v16b bf = load_frag(Wd + (size_t)(g * H_ + ncol) * HP + k0, HP);
        acc[0][g] = wmma_bf(a0, bf, acc[0][g]);
        acc[1][g] = wmma_bf(a1, bf, acc[1][g]);
      }
    }
    __syncthreads();  // all h reads done before updates
#pragma unroll
    for (int mt = 0; mt < 2; ++mt)
#pragma unroll
      for (int r = 0; r < 8; ++r) {
        int gm = mt * 16 + hi8 + r;
        float si = 1.f / (1.f + __expf(-acc[mt][0][r]));
        float sf = 1.f / (1.f + __expf(-acc[mt][1][r]));
        float tg = tanhf(acc[mt][2][r]);
        float so = 1.f / (1.f + __expf(-acc[mt][3][r]));
        float cn = sf * cst[mt][r] + si * tg;
        float hn = so * tanhf(cn);
        bool  m  = t < len[mt][r];
        float hp = (float)h_lds[(size_t)gm * HP + ncol + nl];
        cst[mt][r] = m ? cn : cst[mt][r];
        float h2 = m ? hn : hp;
        h_lds[(size_t)gm * HP + ncol + nl] = (__bf16)h2;
        ycat[((size_t)t * B_ + gm) * DC + dir * H_ + ncol + nl] = (__bf16)(m ? hn : 0.f);
      }
    __syncthreads();
  }
}

// ---------------------------------------------------------------- biaffine arc
// block=(b, xtile of 32 x's), 256 threads. u = xd_tile @ WarcT^T (bf16 in LDS),
// s_arc[b,x,y] = u @ arc_h[b]^T. Time-major rows: row(x,b)=x*B_+b, stride B_*AP.
// Both K loops use the ping-pong pipeline (AP/32 = 16 steps, even).
__global__ void __launch_bounds__(256)
s_arc_kernel(const __bf16* __restrict__ xd,     // [TB][AP]
             const __bf16* __restrict__ WarcT,  // [AP][AP]
             const __bf16* __restrict__ ah,     // [TB][AP]
             float* __restrict__ sarc) {        // [B][T][T]
  __shared__ __bf16 u_lds[32 * AP];
  const int b = blockIdx.x, xt = blockIdx.y;
  const int lane = threadIdx.x & 31, wave = threadIdx.x >> 5;
  const int nl = lane & 15, hi8 = (lane >> 4) * 8;
  const int strideX = B_ * AP;
  const int x0 = xt * 32;
  const __bf16* Ar0 = xd + ((size_t)x0 * B_ + b) * AP;
  const __bf16* Ar1 = xd + ((size_t)(x0 + 16) * B_ + b) * AP;
  const __bf16* Wr  = WarcT + (size_t)(wave * 64) * AP;

  v8f acc[2][4];
#pragma unroll
  for (int mt = 0; mt < 2; ++mt)
#pragma unroll
    for (int j = 0; j < 4; ++j) acc[mt][j] = vzero8();
  {
    v16b a0A = load_frag(Ar0, strideX);
    v16b a1A = load_frag(Ar1, strideX);
    v16b bfA[4], bfB[4];
#pragma unroll
    for (int j = 0; j < 4; ++j) bfA[j] = load_frag(Wr + (size_t)(16 * j) * AP, AP);
    for (int s = 1; s + 1 < (AP >> 5); s += 2) {
      v16b a0B = load_frag(Ar0 + s * 32, strideX);
      v16b a1B = load_frag(Ar1 + s * 32, strideX);
#pragma unroll
      for (int j = 0; j < 4; ++j)
        bfB[j] = load_frag(Wr + (size_t)(16 * j) * AP + s * 32, AP);
#pragma unroll
      for (int j = 0; j < 4; ++j) {
        acc[0][j] = wmma_bf(a0A, bfA[j], acc[0][j]);
        acc[1][j] = wmma_bf(a1A, bfA[j], acc[1][j]);
      }
      a0A = load_frag(Ar0 + (s + 1) * 32, strideX);
      a1A = load_frag(Ar1 + (s + 1) * 32, strideX);
#pragma unroll
      for (int j = 0; j < 4; ++j)
        bfA[j] = load_frag(Wr + (size_t)(16 * j) * AP + (s + 1) * 32, AP);
#pragma unroll
      for (int j = 0; j < 4; ++j) {
        acc[0][j] = wmma_bf(a0B, bfB[j], acc[0][j]);
        acc[1][j] = wmma_bf(a1B, bfB[j], acc[1][j]);
      }
    }
    {  // last prefetched pair (S=16 even -> s==15 here)
      const int s = (AP >> 5) - 1;
      v16b a0B = load_frag(Ar0 + s * 32, strideX);
      v16b a1B = load_frag(Ar1 + s * 32, strideX);
#pragma unroll
      for (int j = 0; j < 4; ++j)
        bfB[j] = load_frag(Wr + (size_t)(16 * j) * AP + s * 32, AP);
#pragma unroll
      for (int j = 0; j < 4; ++j) {
        acc[0][j] = wmma_bf(a0A, bfA[j], acc[0][j]);
        acc[1][j] = wmma_bf(a1A, bfA[j], acc[1][j]);
      }
#pragma unroll
      for (int j = 0; j < 4; ++j) {
        acc[0][j] = wmma_bf(a0B, bfB[j], acc[0][j]);
        acc[1][j] = wmma_bf(a1B, bfB[j], acc[1][j]);
      }
    }
  }
#pragma unroll
  for (int mt = 0; mt < 2; ++mt)
#pragma unroll
    for (int j = 0; j < 4; ++j)
#pragma unroll
      for (int r = 0; r < 8; ++r)
        u_lds[(size_t)(mt * 16 + hi8 + r) * AP + wave * 64 + 16 * j + nl] =
            (__bf16)acc[mt][j][r];
  __syncthreads();

  v8f s2[2][2];
#pragma unroll
  for (int mt = 0; mt < 2; ++mt)
#pragma unroll
    for (int jj = 0; jj < 2; ++jj) s2[mt][jj] = vzero8();
  {
    const __bf16* Brow0 = ah + ((size_t)(wave * 2 + 0) * 16 * B_ + b) * AP;
    const __bf16* Brow1 = ah + ((size_t)(wave * 2 + 1) * 16 * B_ + b) * AP;
    v16b a0A = load_frag(u_lds, AP);
    v16b a1A = load_frag(u_lds + 16 * AP, AP);
    v16b b0A = load_frag(Brow0, strideX);
    v16b b1A = load_frag(Brow1, strideX);
    for (int s = 1; s + 1 < (AP >> 5); s += 2) {
      v16b a0B = load_frag(u_lds + s * 32, AP);
      v16b a1B = load_frag(u_lds + 16 * AP + s * 32, AP);
      v16b b0B = load_frag(Brow0 + s * 32, strideX);
      v16b b1B = load_frag(Brow1 + s * 32, strideX);
      s2[0][0] = wmma_bf(a0A, b0A, s2[0][0]);
      s2[1][0] = wmma_bf(a1A, b0A, s2[1][0]);
      s2[0][1] = wmma_bf(a0A, b1A, s2[0][1]);
      s2[1][1] = wmma_bf(a1A, b1A, s2[1][1]);
      a0A = load_frag(u_lds + (s + 1) * 32, AP);
      a1A = load_frag(u_lds + 16 * AP + (s + 1) * 32, AP);
      b0A = load_frag(Brow0 + (s + 1) * 32, strideX);
      b1A = load_frag(Brow1 + (s + 1) * 32, strideX);
      s2[0][0] = wmma_bf(a0B, b0B, s2[0][0]);
      s2[1][0] = wmma_bf(a1B, b0B, s2[1][0]);
      s2[0][1] = wmma_bf(a0B, b1B, s2[0][1]);
      s2[1][1] = wmma_bf(a1B, b1B, s2[1][1]);
    }
    {
      const int s = (AP >> 5) - 1;
      v16b a0B = load_frag(u_lds + s * 32, AP);
      v16b a1B = load_frag(u_lds + 16 * AP + s * 32, AP);
      v16b b0B = load_frag(Brow0 + s * 32, strideX);
      v16b b1B = load_frag(Brow1 + s * 32, strideX);
      s2[0][0] = wmma_bf(a0A, b0A, s2[0][0]);
      s2[1][0] = wmma_bf(a1A, b0A, s2[1][0]);
      s2[0][1] = wmma_bf(a0A, b1A, s2[0][1]);
      s2[1][1] = wmma_bf(a1A, b1A, s2[1][1]);
      s2[0][0] = wmma_bf(a0B, b0B, s2[0][0]);
      s2[1][0] = wmma_bf(a1B, b0B, s2[1][0]);
      s2[0][1] = wmma_bf(a0B, b1B, s2[0][1]);
      s2[1][1] = wmma_bf(a1B, b1B, s2[1][1]);
    }
  }
#pragma unroll
  for (int mt = 0; mt < 2; ++mt)
#pragma unroll
    for (int jj = 0; jj < 2; ++jj)
#pragma unroll
      for (int r = 0; r < 8; ++r) {
        int x = x0 + mt * 16 + hi8 + r;
        int y = (wave * 2 + jj) * 16 + nl;
        sarc[((size_t)b * T_ + x) * T_ + y] = s2[mt][jj][r];
      }
}

// ---------------------------------------------------------------- biaffine rel
// block=(b, xtile of 32), loop o over 37 rel matrices; W_rel[o]^T staged in LDS.
__global__ void __launch_bounds__(256)
s_rel_kernel(const __bf16* __restrict__ rd,     // [TB][RP]
             const __bf16* __restrict__ WrelT,  // [NRELS][RP][RP]
             const __bf16* __restrict__ rh,     // [TB][RP]
             float* __restrict__ srel) {        // [B][T][T][NRELS]
  __shared__ __bf16 w_lds[RP * RP];
  __shared__ __bf16 u_lds[32 * RP];
  const int b = blockIdx.x, xt = blockIdx.y;
  const int lane = threadIdx.x & 31, wave = threadIdx.x >> 5;
  const int nl = lane & 15, hi8 = (lane >> 4) * 8;
  const int strideX = B_ * RP;
  const int x0 = xt * 32;

  // hoist rd fragments (constant across o)
  v16b ra0[4], ra1[4];
#pragma unroll
  for (int kk = 0; kk < 4; ++kk) {
    ra0[kk] = load_frag(rd + ((size_t)x0 * B_ + b) * RP + kk * 32, strideX);
    ra1[kk] = load_frag(rd + ((size_t)(x0 + 16) * B_ + b) * RP + kk * 32, strideX);
  }

  for (int o = 0; o < NRELS; ++o) {
    __syncthreads();  // prev iteration fully consumed w_lds/u_lds
    {
      const v8b* src = (const v8b*)(WrelT + (size_t)o * RP * RP);
      for (int e = threadIdx.x; e < (RP * RP) / 8; e += blockDim.x)
        ((v8b*)w_lds)[e] = src[e];
      if (o + 1 < NRELS)
        __builtin_prefetch(WrelT + (size_t)(o + 1) * RP * RP, 0, 0);
    }
    __syncthreads();
    // stage 1: u = rd_tile @ W^T; wave owns n-tile = wave
    v8f acc[2];
    acc[0] = vzero8(); acc[1] = vzero8();
#pragma unroll
    for (int kk = 0; kk < 4; ++kk) {
      v16b bf = load_frag(w_lds + (size_t)(wave * 16) * RP + kk * 32, RP);
      acc[0] = wmma_bf(ra0[kk], bf, acc[0]);
      acc[1] = wmma_bf(ra1[kk], bf, acc[1]);
    }
#pragma unroll
    for (int mt = 0; mt < 2; ++mt)
#pragma unroll
      for (int r = 0; r < 8; ++r)
        u_lds[(size_t)(mt * 16 + hi8 + r) * RP + wave * 16 + nl] = (__bf16)acc[mt][r];
    __syncthreads();
    // stage 2: s = u @ rh[b]^T; wave owns 2 y-tiles
    v8f s[2][2];
#pragma unroll
    for (int mt = 0; mt < 2; ++mt) { s[mt][0] = vzero8(); s[mt][1] = vzero8(); }
#pragma unroll
    for (int kk = 0; kk < 4; ++kk) {
      v16b a0 = load_frag(u_lds + kk * 32, RP);
      v16b a1 = load_frag(u_lds + 16 * RP + kk * 32, RP);
#pragma unroll
      for (int jj = 0; jj < 2; ++jj) {
        int yb = (wave * 2 + jj) * 16;
        v16b bf = load_frag(rh + ((size_t)yb * B_ + b) * RP + kk * 32, strideX);
        s[0][jj] = wmma_bf(a0, bf, s[0][jj]);
        s[1][jj] = wmma_bf(a1, bf, s[1][jj]);
      }
    }
#pragma unroll
    for (int mt = 0; mt < 2; ++mt)
#pragma unroll
      for (int jj = 0; jj < 2; ++jj)
#pragma unroll
        for (int r = 0; r < 8; ++r) {
          int x = x0 + mt * 16 + hi8 + r;
          int y = (wave * 2 + jj) * 16 + nl;
          srel[(((size_t)b * T_ + x) * T_ + y) * NRELS + o] = s[mt][jj][r];
        }
  }
}

// ---------------------------------------------------------------- host
extern "C" void kernel_launch(void* const* d_in, const int* in_sizes, int n_in,
                              void* d_out, int out_size, void* d_ws, size_t ws_size,
                              hipStream_t stream) {
  const float* x       = (const float*)d_in[0];
  const int*   lengths = (const int*)d_in[1];
  const float* w_ih0   = (const float*)d_in[2];
  const float* w_hh0   = (const float*)d_in[3];
  const float* b0      = (const float*)d_in[4];
  const float* w_ih_r  = (const float*)d_in[5];
  const float* w_hh_r  = (const float*)d_in[6];
  const float* b_r     = (const float*)d_in[7];
  const float* w_arc_d = (const float*)d_in[8];
  const float* b_arc_d = (const float*)d_in[9];
  const float* w_arc_h = (const float*)d_in[10];
  const float* b_arc_h = (const float*)d_in[11];
  const float* w_rel_d = (const float*)d_in[12];
  const float* b_rel_d = (const float*)d_in[13];
  const float* w_rel_h = (const float*)d_in[14];
  const float* b_rel_h = (const float*)d_in[15];
  const float* W_arc   = (const float*)d_in[16];
  const float* W_rel   = (const float*)d_in[17];
  float* out = (float*)d_out;

  char* base = (char*)d_ws;
  size_t off = 0;
  auto alloc = [&](size_t bytes) -> char* {
    char* p = base + off;
    off = (off + bytes + 255) & ~(size_t)255;
    return p;
  };
  __bf16* xs    = (__bf16*)alloc((size_t)TB * D_ * 2);
  __bf16* catA  = (__bf16*)alloc((size_t)TB * DC * 2);
  __bf16* catB  = (__bf16*)alloc((size_t)TB * DC * 2);
  __bf16* Pbuf  = (__bf16*)alloc((size_t)2 * TB * G4 * 2);
  __bf16* wih0b = (__bf16*)alloc((size_t)2 * G4 * D_ * 2);
  __bf16* wihrb = (__bf16*)alloc((size_t)4 * G4 * DC * 2);
  __bf16* whhb  = (__bf16*)alloc((size_t)6 * G4 * HP * 2);
  __bf16* wadb  = (__bf16*)alloc((size_t)AP * DC * 2);   // row-padded to AP
  __bf16* wahb  = (__bf16*)alloc((size_t)AP * DC * 2);
  __bf16* wrdb  = (__bf16*)alloc((size_t)RP * DC * 2);   // row-padded to RP
  __bf16* wrhb  = (__bf16*)alloc((size_t)RP * DC * 2);
  __bf16* adp   = (__bf16*)alloc((size_t)TB * AP * 2);
  __bf16* ahp   = (__bf16*)alloc((size_t)TB * AP * 2);
  __bf16* rdp   = (__bf16*)alloc((size_t)TB * RP * 2);
  __bf16* rhp   = (__bf16*)alloc((size_t)TB * RP * 2);
  __bf16* warcT = (__bf16*)alloc((size_t)AP * AP * 2);
  __bf16* wrelT = (__bf16*)alloc((size_t)NRELS * RP * RP * 2);
  (void)ws_size; (void)in_sizes; (void)n_in; (void)out_size;

  auto gsz = [](size_t n) {
    size_t b = (n + 255) / 256;
    return (unsigned)(b > 4096 ? 4096 : b);
  };
  auto cast = [&](const float* s, __bf16* d, size_t n) {
    cast_bf16_kernel<<<gsz(n), 256, 0, stream>>>(s, d, n);
  };

  // pack / cast
  pack_x_kernel<<<gsz((size_t)TB * D_), 256, 0, stream>>>(x, xs);
  cast(w_ih0, wih0b, (size_t)2 * G4 * D_);
  cast(w_ih_r, wihrb, (size_t)4 * G4 * DC);
  pack_whh_kernel<<<gsz((size_t)2 * G4 * HP), 256, 0, stream>>>(w_hh0, whhb, 2);
  pack_whh_kernel<<<gsz((size_t)4 * G4 * HP), 256, 0, stream>>>(w_hh_r, whhb + (size_t)2 * G4 * HP, 4);
  pack_padrows_kernel<<<gsz((size_t)AP * DC), 256, 0, stream>>>(w_arc_d, wadb, A_, AP, DC);
  pack_padrows_kernel<<<gsz((size_t)AP * DC), 256, 0, stream>>>(w_arc_h, wahb, A_, AP, DC);
  pack_padrows_kernel<<<gsz((size_t)RP * DC), 256, 0, stream>>>(w_rel_d, wrdb, R_, RP, DC);
  pack_padrows_kernel<<<gsz((size_t)RP * DC), 256, 0, stream>>>(w_rel_h, wrhb, R_, RP, DC);
  pack_warcT_kernel<<<gsz((size_t)AP * AP), 256, 0, stream>>>(W_arc, warcT);
  pack_wrelT_kernel<<<gsz((size_t)NRELS * RP * RP), 256, 0, stream>>>(W_rel, wrelT);
  init_pad_kernel<<<gsz((size_t)TB * AP), 256, 0, stream>>>(adp, TB, AP, A_);
  init_pad_kernel<<<gsz((size_t)TB * AP), 256, 0, stream>>>(ahp, TB, AP, -1);
  init_pad_kernel<<<gsz((size_t)TB * RP), 256, 0, stream>>>(rdp, TB, RP, R_);
  init_pad_kernel<<<gsz((size_t)TB * RP), 256, 0, stream>>>(rhp, TB, RP, R_);

  auto gemm = [&](const __bf16* A, int lda, const __bf16* Bw, int ldb, int N, int K,
                  const float* bias, __bf16* o, int ldo, int ocol, int act) {
    dim3 g(TB / 128, (unsigned)((N + 63) / 64));
    gemm_bf16_kernel<<<g, 256, 0, stream>>>(A, lda, Bw, ldb, N, K, bias, o, ldo, ocol, act);
  };

  // BiLSTM: 3 layers
  const __bf16* in = xs;
  int Kin = D_;
  for (int layer = 0; layer < 3; ++layer) {
    const __bf16* wih  = (layer == 0) ? wih0b : wihrb + (size_t)(layer - 1) * 2 * G4 * DC;
    const float*  bias = (layer == 0) ? b0    : b_r + (size_t)(layer - 1) * 2 * G4;
    for (int dir = 0; dir < 2; ++dir)
      gemm(in, Kin, wih + (size_t)dir * G4 * Kin, Kin, G4, Kin,
           bias + (size_t)dir * G4, Pbuf + (size_t)dir * TB * G4, G4, 0, 0);
    __bf16* ocat = (layer == 1) ? catB : catA;
    lstm_layer_kernel<<<2, 800, 0, stream>>>(Pbuf, whhb + (size_t)layer * 2 * G4 * HP,
                                             lengths, ocat);
    in = ocat;
    Kin = DC;
  }
  const __bf16* feats = catA;  // layer2 output

  // MLP heads (leaky relu 0.1) into padded buffers; weight rows zero-padded so
  // fragment loads are branch-free, store guard keeps the `ones` pad columns.
  gemm(feats, DC, wadb, DC, A_, DC, b_arc_d, adp, AP, 0, 1);
  gemm(feats, DC, wahb, DC, A_, DC, b_arc_h, ahp, AP, 0, 1);
  gemm(feats, DC, wrdb, DC, R_, DC, b_rel_d, rdp, RP, 0, 1);
  gemm(feats, DC, wrhb, DC, R_, DC, b_rel_h, rhp, RP, 0, 1);

  // biaffine scores
  float* sarc = out;                         // B*T*T
  float* srel = out + (size_t)B_ * T_ * T_;  // B*T*T*NRELS
  s_arc_kernel<<<dim3(B_, T_ / 32), 256, 0, stream>>>(adp, warcT, ahp, sarc);
  s_rel_kernel<<<dim3(B_, T_ / 32), 256, 0, stream>>>(rdp, wrelT, rhp, srel);
}
